// NTXentLoss_84318797955094
// MI455X (gfx1250) — matrix operand order repaired and checked by reading, compiled, not run on hardware
//
#include <hip/hip_runtime.h>
#include <hip/hip_bf16.h>

typedef __attribute__((ext_vector_type(16))) _Float16 v16h;
typedef __attribute__((ext_vector_type(8)))  _Float16 v8h;
typedef __attribute__((ext_vector_type(8)))  float    v8f;
typedef __attribute__((ext_vector_type(4)))  float    v4f;

#define BATCH 4096
#define DIM   256
#define INV_T 2.0f      /* 1/TEMP, TEMP = 0.5 */
#define BPAD  264       /* LDS row stride in halves: 256 + 8 (bank-rotating pad) */

// ---------------------------------------------------------------------------
// Kernel 0: zero the fp32 accumulators (rowsum[4096] | colsum[4096] | diag[1])
// ---------------------------------------------------------------------------
__global__ void ntx_zero(float* __restrict__ sums, int n) {
    int i = blockIdx.x * blockDim.x + threadIdx.x;
    if (i < n) sums[i] = 0.0f;
}

// ---------------------------------------------------------------------------
// Kernel 1: L2-normalize rows of z_i / z_j (fp32) and store as fp16.
// One wave per row: 32 lanes x 8 elements = 256.
// ---------------------------------------------------------------------------
__global__ __launch_bounds__(256) void ntx_normalize(
    const float* __restrict__ zi, const float* __restrict__ zj,
    _Float16* __restrict__ A, _Float16* __restrict__ B) {
    const int wave = threadIdx.x >> 5;
    const int lane = threadIdx.x & 31;
    const int row  = blockIdx.x * 8 + wave;              // 0 .. 8191

    const float* src;
    _Float16*    dst;
    if (row < BATCH) {
        src = zi + (size_t)row * DIM;
        dst = A  + (size_t)row * DIM;
    } else {
        src = zj + (size_t)(row - BATCH) * DIM;
        dst = B  + (size_t)(row - BATCH) * DIM;
    }

    v4f x0 = *(const v4f*)(src + lane * 8);
    v4f x1 = *(const v4f*)(src + lane * 8 + 4);

    float ss = 0.0f;
#pragma unroll
    for (int i = 0; i < 4; ++i) ss += x0[i] * x0[i] + x1[i] * x1[i];
    ss += __shfl_xor(ss, 1);
    ss += __shfl_xor(ss, 2);
    ss += __shfl_xor(ss, 4);
    ss += __shfl_xor(ss, 8);
    ss += __shfl_xor(ss, 16);

    const float nrm = sqrtf(ss);
    const float scl = 1.0f / fmaxf(nrm, 1e-12f);

    v8h h;
#pragma unroll
    for (int i = 0; i < 4; ++i) {
        h[i]     = (_Float16)(x0[i] * scl);
        h[4 + i] = (_Float16)(x1[i] * scl);
    }
    *(v8h*)(dst + lane * 8) = h;
}

// ---------------------------------------------------------------------------
// Kernel 2: fused S = A · Bᵀ (4096x4096x256, fp16 WMMA, fp32 accum) with
// exp(S/T) row/col reductions and diagonal capture in the epilogue.
//
// Block = 256 threads (8 waves) computes a 128(row) x 64(col) tile of S.
// The block's 64-row B panel (32 KB) is staged once into LDS via
// GLOBAL_LOAD_ASYNC_TO_LDS_B128 (ASYNCcnt), then each wave runs 8 K-steps of
// V_WMMA_F32_16X16X32_F16 reading B fragments from LDS (ds_load_b128) and its
// private A fragments from global.
// ---------------------------------------------------------------------------
__global__ __launch_bounds__(256) void ntx_gemm_fused(
    const _Float16* __restrict__ A, const _Float16* __restrict__ B,
    float* __restrict__ rowsum, float* __restrict__ colsum,
    float* __restrict__ diagsum) {
    __shared__ __align__(16) _Float16 Bs[64 * BPAD];   // 33,792 bytes

    const int lane = threadIdx.x & 31;
    const int wave = threadIdx.x >> 5;
    const int lrow = lane & 15;          // row/col within a 16-wide fragment
    const int hi   = lane >> 4;          // lane half (K-phase selector)

    const int m0 = blockIdx.y * 128 + wave * 16;   // output row base
    const int n0 = blockIdx.x * 64;                // output col base

    // ---- stage the 64-row B panel into LDS with async copies ----
    // Global panel is contiguous: 64 rows x 512 B = 2048 b128 chunks.
    // Each of the 256 threads issues 8 GLOBAL_LOAD_ASYNC_TO_LDS_B128.
    {
        const unsigned lds_base =
            (unsigned)(unsigned long long)(void*)&Bs[0];   // LDS byte offset
        const unsigned long long gbase =
            (unsigned long long)(const void*)(B + (size_t)n0 * DIM);
#pragma unroll
        for (int i = 0; i < 8; ++i) {
            const unsigned chunk  = (unsigned)threadIdx.x + 256u * i; // 0..2047
            const unsigned row    = chunk >> 5;                       // /32
            const unsigned within = chunk & 31u;                      // %32
            const unsigned ldsoff = lds_base + row * (BPAD * 2u) + within * 16u;
            const unsigned long long gaddr = gbase + (unsigned long long)chunk * 16ull;
            asm volatile("global_load_async_to_lds_b128 %0, %1, off"
                         :: "v"(ldsoff), "v"(gaddr) : "memory");
        }
        asm volatile("s_wait_asynccnt 0x0" ::: "memory");
        __syncthreads();
    }

    // A fragment (16x32 fp16): lane l holds row (l&15);
    //   lanes 0-15: K = 0..15; lanes 16-31: K = 16..31
    const _Float16* arow = A + (size_t)(m0 + lrow) * DIM + hi * 16;

    // B fragment (32x16 fp16): B[k][n] = Zj[n][k]; lane l holds col (l&15);
    //   lanes 0-15: K = 0..15; lanes 16-31: K = 16..31 (contiguous in Zj row)
    const _Float16* bs0 = &Bs[(0  + lrow) * BPAD + hi * 16];
    const _Float16* bs1 = &Bs[(16 + lrow) * BPAD + hi * 16];
    const _Float16* bs2 = &Bs[(32 + lrow) * BPAD + hi * 16];
    const _Float16* bs3 = &Bs[(48 + lrow) * BPAD + hi * 16];
    const _Float16* bsrows[4] = {bs0, bs1, bs2, bs3};

    v8f acc[4] = {};

#pragma unroll
    for (int k = 0; k < DIM; k += 32) {
        v8h a0 = *(const v8h*)(arow + k);
        v8h a1 = *(const v8h*)(arow + k + 8);
        v16h a = __builtin_shufflevector(a0, a1, 0, 1, 2, 3, 4, 5, 6, 7,
                                                 8, 9, 10, 11, 12, 13, 14, 15);
#pragma unroll
        for (int t = 0; t < 4; ++t) {
            v8h b0 = *(const v8h*)(bsrows[t] + k);       // ds_load_b128
            v8h b1 = *(const v8h*)(bsrows[t] + k + 8);   // ds_load_b128
            v16h b = __builtin_shufflevector(b0, b1, 0, 1, 2, 3, 4, 5, 6, 7,
                                                     8, 9, 10, 11, 12, 13, 14, 15);
            acc[t] = __builtin_amdgcn_wmma_f32_16x16x32_f16(
                false, a, false, b, (short)0, acc[t], false, false);
        }
    }

    // ---- fused epilogue: p = exp(s/T); row sums, col sums, diagonal ----
    // C/D layout: lane l, reg r -> row = m0 + r + 8*(l>=16), col = n0+16t+(l&15)
    float rowp[8] = {0, 0, 0, 0, 0, 0, 0, 0};
    float sdiag   = 0.0f;
    bool  ondiag  = false;

#pragma unroll
    for (int t = 0; t < 4; ++t) {
        const int col = n0 + 16 * t + lrow;
        float colp = 0.0f;
#pragma unroll
        for (int r = 0; r < 8; ++r) {
            const float s = acc[t][r];
            const float p = __expf(s * INV_T);
            rowp[r] += p;
            colp    += p;
            const int row = m0 + r + hi * 8;
            if (row == col) { sdiag += s; ondiag = true; }
        }
        colp += __shfl_xor(colp, 16);        // fold the two row-halves
        if (hi == 0) atomicAdd(&colsum[col], colp);
    }
    if (ondiag) atomicAdd(diagsum, sdiag);

#pragma unroll
    for (int r = 0; r < 8; ++r) {
        float v = rowp[r];
        v += __shfl_xor(v, 1);
        v += __shfl_xor(v, 2);
        v += __shfl_xor(v, 4);
        v += __shfl_xor(v, 8);               // reduce the 16 columns
        if (lrow == 0) atomicAdd(&rowsum[m0 + r + hi * 8], v);
    }
}

// ---------------------------------------------------------------------------
// Kernel 3: loss = (Σ log rowsum + Σ log colsum − 2·diag/T) / (2B)
// ---------------------------------------------------------------------------
__global__ __launch_bounds__(256) void ntx_finalize(
    const float* __restrict__ rowsum, const float* __restrict__ colsum,
    const float* __restrict__ diagsum, float* __restrict__ out) {
    __shared__ float red[256];
    float acc = 0.0f;
    for (int i = threadIdx.x; i < BATCH; i += 256)
        acc += __logf(rowsum[i]) + __logf(colsum[i]);
    red[threadIdx.x] = acc;
    __syncthreads();
    for (int s = 128; s > 0; s >>= 1) {
        if (threadIdx.x < s) red[threadIdx.x] += red[threadIdx.x + s];
        __syncthreads();
    }
    if (threadIdx.x == 0) {
        const float total = red[0] - 2.0f * INV_T * diagsum[0];
        out[0] = total / (2.0f * (float)BATCH);
    }
}

// ---------------------------------------------------------------------------
extern "C" void kernel_launch(void* const* d_in, const int* in_sizes, int n_in,
                              void* d_out, int out_size, void* d_ws, size_t ws_size,
                              hipStream_t stream) {
    (void)in_sizes; (void)n_in; (void)out_size; (void)ws_size;
    const float* zi = (const float*)d_in[0];
    const float* zj = (const float*)d_in[1];
    float* out = (float*)d_out;

    char* ws = (char*)d_ws;
    _Float16* A = (_Float16*)ws;                                   // 2 MB
    _Float16* B = (_Float16*)(ws + (size_t)BATCH * DIM * 2);       // 2 MB
    float* sums = (float*)(ws + (size_t)2 * BATCH * DIM * 2);      // 32 KB + 4
    float* rowsum  = sums;
    float* colsum  = sums + BATCH;
    float* diagsum = sums + 2 * BATCH;

    const int nsum = 2 * BATCH + 1;
    ntx_zero<<<(nsum + 255) / 256, 256, 0, stream>>>(sums, nsum);

    ntx_normalize<<<(2 * BATCH) / 8, 256, 0, stream>>>(zi, zj, A, B);

    dim3 grid(BATCH / 64, BATCH / 128);   // (64, 32) blocks of 256 threads
    ntx_gemm_fused<<<grid, 256, 0, stream>>>(A, B, rowsum, colsum, diagsum);

    ntx_finalize<<<1, 256, 0, stream>>>(rowsum, colsum, diagsum, out);
}
